// GCN_Layer_678604833479
// MI455X (gfx1250) — matrix operand order repaired and checked by reading, not compiled, not run on hardware
//
#include <hip/hip_runtime.h>

// GCN layer: out = leaky_relu((I - D^.5 A D^-.5) @ (BN(H) @ W + b))
// B=32, L=1024, C=64, OUT=64. All f32 I/O; big GEMM via v_wmma_f32_16x16x32_f16.

typedef __attribute__((ext_vector_type(16))) _Float16 v16h;
typedef __attribute__((ext_vector_type(8)))  float    v8f;

union AFrag { v16h v; _Float16 h[16]; };
union CFrag { v8f  v; float    f[8];  };

#define NROWS 32768   // B*L
#define BN_EPS 1e-5f

// ---------------- Kernel 1a: per-block partial BN sums (deterministic) -------
__global__ __launch_bounds__(256)
void bn_partial(const float* __restrict__ H, float* __restrict__ sumP,
                float* __restrict__ sqP) {
    __shared__ float s1[256], s2[256];
    const int t = threadIdx.x;
    const int c = t & 63;          // channel
    const int rs = t >> 6;         // row sub-lane 0..3
    const size_t row0 = (size_t)blockIdx.x * 128 + rs;
    float a = 0.f, q = 0.f;
    for (int i = 0; i < 32; ++i) {            // 128 rows per block, stride 4
        float v = H[(row0 + (size_t)i * 4) * 64 + c];
        a += v; q += v * v;
    }
    s1[t] = a; s2[t] = q;
    __syncthreads();
    if (t < 64) {
        a = s1[t] + s1[t + 64] + s1[t + 128] + s1[t + 192];
        q = s2[t] + s2[t + 64] + s2[t + 128] + s2[t + 192];
        sumP[blockIdx.x * 64 + t] = a;        // t == channel here
        sqP [blockIdx.x * 64 + t] = q;
    }
}

// ---------------- Kernel 1b: finalize BN scale/bias --------------------------
__global__ void bn_finish(const float* __restrict__ sumP, const float* __restrict__ sqP,
                          const float* __restrict__ gamma, const float* __restrict__ beta,
                          float* __restrict__ bnS, float* __restrict__ bnB) {
    const int c = threadIdx.x;
    if (c < 64) {
        float a = 0.f, q = 0.f;
        for (int i = 0; i < 256; ++i) { a += sumP[i * 64 + c]; q += sqP[i * 64 + c]; }
        const float inv = 1.f / (float)NROWS;
        const float mean = a * inv;
        const float var  = q * inv - mean * mean;   // biased variance
        const float sc   = gamma[c] * rsqrtf(var + BN_EPS);
        bnS[c] = sc;
        bnB[c] = beta[c] - mean * sc;
    }
}

// ---------------- Kernel 2: row degrees -> sqrt(d), rsqrt(d) -----------------
__global__ __launch_bounds__(256)
void degrees(const float* __restrict__ A, float* __restrict__ sqd,
             float* __restrict__ rsd) {
    const int lane = threadIdx.x & 31;
    const int w    = threadIdx.x >> 5;
    const size_t row = (size_t)blockIdx.x * 8 + w;     // 0..32767
    const float* p = A + row * 1024;
    float s = 0.f;
    #pragma unroll 8
    for (int i = 0; i < 32; ++i) s += p[lane + (i << 5)];   // coalesced
    #pragma unroll
    for (int off = 16; off; off >>= 1) s += __shfl_xor(s, off, 32);
    if (lane == 0) { sqd[row] = sqrtf(s); rsd[row] = rsqrtf(s); }
}

// ---------------- Kernel 3: X = BN(H) @ W + b; store f32 + WMMA-B f16 frags --
__global__ __launch_bounds__(256)
void gcn_x(const float* __restrict__ H, const float* __restrict__ W,
           const float* __restrict__ bias, const float* __restrict__ bnS,
           const float* __restrict__ bnB, float* __restrict__ Xf32,
           _Float16* __restrict__ Xh) {
    __shared__ float sW[64 * 64];
    __shared__ float sH[4 * 64];
    const int t = threadIdx.x;
    for (int i = t; i < 4096; i += 256) sW[i] = W[i];
    const int rs = t >> 6;                 // row within block (0..3)
    const int o  = t & 63;                 // output channel
    const size_t g = (size_t)blockIdx.x * 4 + rs;     // global node row
    sH[(rs << 6) + o] = H[g * 64 + o] * bnS[o] + bnB[o];   // fused BN
    __syncthreads();
    float s = bias[o];
    #pragma unroll
    for (int c = 0; c < 64; ++c) s += sH[(rs << 6) + c] * sW[(c << 6) + o];
    Xf32[g * 64 + o] = s;
    // Scatter into this code's B-fragment order (16x16x32 f16, contiguous k, same as A):
    //   lanes 0-15: N=lane,   K=0..15 ; lanes 16-31: N=lane-16, K=16..31
    const int b  = (int)(g >> 10), l = (int)(g & 1023);
    const int kc = l >> 5, kk = l & 31;
    const int hi = kk >> 4, e = kk & 15;
    const int nt = o >> 4,  nn = o & 15;
    const int lane = (hi << 4) + nn;
    Xh[((((size_t)(b * 32 + kc) * 4 + nt) * 32 + lane) << 4) + e] = (_Float16)s;
}

// ---------------- Kernel 4: out = leaky(Xf32 - A_norm @ X)  [WMMA f16] -------
// One wave per (batch, 16-row m-tile). 4 f32 accumulators cover all 64 cols;
// the A fragment is loaded once per 32-wide K-chunk and reused for 4 WMMAs.
__global__ __launch_bounds__(256)
void gcn_main(const float* __restrict__ A, const float* __restrict__ sqd,
              const float* __restrict__ rsd, const float* __restrict__ Xf32,
              const _Float16* __restrict__ Xh, float* __restrict__ out) {
    const int lane = threadIdx.x & 31;
    const int w    = threadIdx.x >> 5;
    const int gw   = blockIdx.x * 8 + w;     // 0..2047
    const int b    = gw >> 6;                // batch
    const int mt   = gw & 63;                // 16-row tile
    const int r    = lane & 15;
    const int hi   = lane >> 4;
    const int row0 = mt * 16;

    const size_t Abase = ((size_t)b << 20) + (size_t)(row0 + r) * 1024;
    const float  sr    = sqd[(b << 10) + row0 + r];      // sqrt(d_row)
    const float* rsdb  = rsd + (b << 10);

    CFrag acc[4];
    #pragma unroll
    for (int nt = 0; nt < 4; ++nt)
        #pragma unroll
        for (int v = 0; v < 8; ++v) acc[nt].f[v] = 0.f;

    const _Float16* xbb = Xh + (((size_t)(b * 32) * 4 * 32 + lane) << 4);

    for (int kc = 0; kc < 32; ++kc) {
        const int k0 = kc << 5;
        const int ka = k0 + 16 * hi;         // A layout: one 16-elem run
        // A tile row (f32) + column rsqrt(d) factors: vector loads, 32B aligned
        float ta[16], tr[16];
        *(float4*)&ta[0]  = *(const float4*)(A + Abase + ka);
        *(float4*)&ta[4]  = *(const float4*)(A + Abase + ka + 4);
        *(float4*)&ta[8]  = *(const float4*)(A + Abase + ka + 8);
        *(float4*)&ta[12] = *(const float4*)(A + Abase + ka + 12);
        *(float4*)&tr[0]  = *(const float4*)(rsdb + ka);
        *(float4*)&tr[4]  = *(const float4*)(rsdb + ka + 4);
        *(float4*)&tr[8]  = *(const float4*)(rsdb + ka + 8);
        *(float4*)&tr[12] = *(const float4*)(rsdb + ka + 12);
        AFrag af;
        #pragma unroll
        for (int e = 0; e < 16; ++e)
            af.h[e] = (_Float16)(sr * ta[e] * tr[e]);   // A_norm element, f16

        // B fragments: contiguous 32B per lane, nt stride = 512 halves
        const _Float16* xb = xbb + ((size_t)kc << 11);  // kc * 4*32*16
        v16h b0 = *(const v16h*)(xb);
        v16h b1 = *(const v16h*)(xb + 512);
        v16h b2 = *(const v16h*)(xb + 1024);
        v16h b3 = *(const v16h*)(xb + 1536);

        acc[0].v = __builtin_amdgcn_wmma_f32_16x16x32_f16(false, af.v, false, b0,
                                                          (short)0, acc[0].v, false, false);
        acc[1].v = __builtin_amdgcn_wmma_f32_16x16x32_f16(false, af.v, false, b1,
                                                          (short)0, acc[1].v, false, false);
        acc[2].v = __builtin_amdgcn_wmma_f32_16x16x32_f16(false, af.v, false, b2,
                                                          (short)0, acc[2].v, false, false);
        acc[3].v = __builtin_amdgcn_wmma_f32_16x16x32_f16(false, af.v, false, b3,
                                                          (short)0, acc[3].v, false, false);
    }

    // Epilogue: C/D layout -> row = row0 + v + 8*hi, col = nt*16 + r
    #pragma unroll
    for (int nt = 0; nt < 4; ++nt) {
        #pragma unroll
        for (int v = 0; v < 8; ++v) {
            const size_t idx = ((size_t)(b << 10) + row0 + v + 8 * hi) * 64 + nt * 16 + r;
            const float val = Xf32[idx] - acc[nt].f[v];
            out[idx] = val >= 0.f ? val : 0.01f * val;
        }
    }
}

// ---------------- Host-side launch ------------------------------------------
extern "C" void kernel_launch(void* const* d_in, const int* in_sizes, int n_in,
                              void* d_out, int out_size, void* d_ws, size_t ws_size,
                              hipStream_t stream) {
    const float* H     = (const float*)d_in[0];
    const float* A     = (const float*)d_in[1];
    const float* gamma = (const float*)d_in[2];
    const float* beta  = (const float*)d_in[3];
    const float* W     = (const float*)d_in[4];
    const float* bias  = (const float*)d_in[5];
    float* out = (float*)d_out;

    float* ws    = (float*)d_ws;
    float* sumP  = ws;                 // 256*64
    float* sqP   = ws + 16384;         // 256*64
    float* bnS   = ws + 32768;         // 64
    float* bnB   = ws + 32832;         // 64
    float* sqd   = ws + 32896;         // 32768
    float* rsd   = ws + 65664;         // 32768
    float* Xf32  = ws + 98432;         // 32768*64
    _Float16* Xh = (_Float16*)(ws + 2195584);   // 32768*64 halves, 32B aligned

    bn_partial<<<256, 256, 0, stream>>>(H, sumP, sqP);
    degrees  <<<4096, 256, 0, stream>>>(A, sqd, rsd);
    bn_finish<<<1, 64, 0, stream>>>(sumP, sqP, gamma, beta, bnS, bnB);
    gcn_x    <<<8192, 256, 0, stream>>>(H, W, bias, bnS, bnB, Xf32, Xh);
    gcn_main <<<256, 256, 0, stream>>>(A, sqd, rsd, Xf32, Xh, out);
}
